// GATEncoder_6717328851290
// MI455X (gfx1250) — matrix operand order repaired and checked by reading, not compiled, not run on hardware
//
#include <hip/hip_runtime.h>
#include <hip/hip_bf16.h>

// ---------------------------------------------------------------------------
// GAT encoder (2 layers) for MI455X / gfx1250, wave32.
//   layer1: h = feat @ W1  (50000x256 @ 256x512) -> WMMA f16->f32
//           edge softmax over dst (4 heads x 128) -> scatter aggregate
//           ELU
//   layer2: h2 = h1 @ W2   (50000x512 @ 512x256) -> WMMA
//           edge softmax (1 head x 256) -> aggregate into d_out
// GEMM: 256 threads (8 waves), block tile 128x64, wave tile 16x64
//       (4 WMMA accumulators per wave), K step 32, LDS-staged operands.
// ---------------------------------------------------------------------------

typedef _Float16 half8  __attribute__((ext_vector_type(8)));
typedef _Float16 half16 __attribute__((ext_vector_type(16)));
typedef float    f32x8  __attribute__((ext_vector_type(8)));

#define GAT_N   50000
#define GAT_E   800000
#define GAT_IND 256
#define GAT_H1  4
#define GAT_D1  128
#define GAT_H2  1
#define GAT_D2  256

#define BLK_M  128
#define BLK_N  64
#define TILE_K 32

// ---------------- f32 -> f16 conversion ----------------
__global__ void k_cvt_f16(const float* __restrict__ in, _Float16* __restrict__ out, int n) {
    int i = blockIdx.x * blockDim.x + threadIdx.x;
    if (i < n) out[i] = (_Float16)in[i];
}

// ---------------- fill u32 (also zeroes f32 buffers) ----------------
__global__ void k_fill_u32(unsigned* __restrict__ p, unsigned v, int n) {
    int i = blockIdx.x * blockDim.x + threadIdx.x;
    if (i < n) p[i] = v;
}

// ---------------- tiled WMMA GEMM: C[MxN] = A[MxK] * B[KxN], f16 in / f32 out
// Requires M % 16 == 0 (tile validity is wave-uniform), N % 64 == 0, K % 32 == 0.
__global__ __launch_bounds__(256) void k_gemm_f16_wmma(
    const _Float16* __restrict__ A, const _Float16* __restrict__ B,
    float* __restrict__ C, int M, int N, int K)
{
    __shared__ _Float16 As[BLK_M * TILE_K];   // 128 x 32 halves = 8 KB (row stride 32)
    __shared__ _Float16 Bt[BLK_N * TILE_K];   // 64 cols x 32 k, transposed: Bt[n*32+k] = 4 KB

    const int tid  = threadIdx.x;
    const int lane = tid & 31;
    const int wave = tid >> 5;                 // 0..7
    const int m0 = blockIdx.x * BLK_M;
    const int n0 = blockIdx.y * BLK_N;

    f32x8 acc[4] = {};

    for (int k0 = 0; k0 < K; k0 += TILE_K) {
        // --- A tile: 128 rows x 32 halves = 256 chunks of 32B, one per thread
        {
            int r   = tid >> 1;                    // 0..127
            int seg = tid & 1;                     // 16 halves each
            int gr  = m0 + r; if (gr >= M) gr = M - 1;   // clamp (stores guarded)
            *(half16*)(As + r * TILE_K + seg * 16) =
                *(const half16*)(A + (size_t)gr * K + k0 + seg * 16);
        }
        // --- B tile (32 x 64), stored transposed into LDS: Bt[n*32+k]
        #pragma unroll
        for (int c = 0; c < 8; ++c) {
            int e  = tid + c * 256;                // 0..2047
            int nn = e & 63, kk = e >> 6;          // adjacent threads -> adjacent n (coalesced)
            Bt[nn * TILE_K + kk] = B[(size_t)(k0 + kk) * N + n0 + nn];
        }
        __syncthreads();

        // A fragment: lane L -> row (L&15); lanes 0-15 carry K 0..15; lanes 16-31 K 16..31
        const int mrow = wave * 16 + (lane & 15);
        const int kg   = lane >> 4;
        half16 a = *(const half16*)(As + mrow * TILE_K + kg * 16);

        // 4 WMMAs per K step: B fragment per 16-col sub-tile (contiguous 32B in Bt)
        #pragma unroll
        for (int t = 0; t < 4; ++t) {
            half16 b = *(const half16*)(Bt + (t * 16 + (lane & 15)) * TILE_K + kg * 16);
            acc[t] = __builtin_amdgcn_wmma_f32_16x16x32_f16(false, a, false, b,
                                                            (short)0, acc[t], false, false);
        }
        __syncthreads();
    }

    // --- store: wave-uniform validity (M is a multiple of 16)
    if (m0 + wave * 16 < M) {
        const int rbase = m0 + wave * 16 + (lane >> 4) * 8;
        #pragma unroll
        for (int t = 0; t < 4; ++t) {
            float* p = C + (size_t)rbase * N + n0 + t * 16 + (lane & 15);
            #pragma unroll
            for (int v = 0; v < 8; ++v) {
                *p = acc[t][v];
                p += N;
            }
        }
    }
}

// ---------------- per-(node,head) attention scalars el/er ----------------
__global__ void k_node_attn(const float* __restrict__ Ht, const float* __restrict__ al,
                            const float* __restrict__ ar, float* __restrict__ el,
                            float* __restrict__ er, int n, int heads, int dim)
{
    int t = blockIdx.x * blockDim.x + threadIdx.x;
    if (t >= n * heads) return;
    int h = t % heads;
    const float* row = Ht + (size_t)t * dim;   // t*dim == node*(heads*dim) + h*dim
    const float* pa  = al + h * dim;
    const float* pb  = ar + h * dim;
    float sl = 0.f, sr = 0.f;
    for (int j = 0; j < dim; j += 4) {
        float4 v  = *(const float4*)(row + j);
        float4 va = *(const float4*)(pa + j);
        float4 vb = *(const float4*)(pb + j);
        sl += v.x * va.x + v.y * va.y + v.z * va.z + v.w * va.w;
        sr += v.x * vb.x + v.y * vb.y + v.z * vb.z + v.w * vb.w;
    }
    el[t] = sl; er[t] = sr;
}

// order-preserving float -> uint mapping for atomic max
__device__ __forceinline__ unsigned orderKey(float x) {
    unsigned u = __float_as_uint(x);
    return (u & 0x80000000u) ? ~u : (u | 0x80000000u);
}

// ---------------- edge pass A: segment max ----------------
__global__ void k_edge_max(const float* __restrict__ el, const float* __restrict__ er,
                           const int* __restrict__ src, const int* __restrict__ dst,
                           unsigned* __restrict__ mu, int E, int H)
{
    int t = blockIdx.x * blockDim.x + threadIdx.x;
    if (t >= E * H) return;
    int e = t / H, h = t - e * H;
    float x = el[src[e] * H + h] + er[dst[e] * H + h];
    x = x > 0.f ? x : 0.2f * x;                       // leaky relu
    atomicMax(&mu[dst[e] * H + h], orderKey(x));
}

// decode max; untouched (no-edge) or non-finite -> 0 (matches reference)
__global__ void k_m_fin(const unsigned* __restrict__ mu, float* __restrict__ mf, int n) {
    int i = blockIdx.x * blockDim.x + threadIdx.x;
    if (i >= n) return;
    unsigned u = mu[i];
    float x = (u & 0x80000000u) ? __uint_as_float(u & 0x7fffffffu)
                                : __uint_as_float(~u);
    unsigned b = __float_as_uint(x);
    if (u == 0u || ((b >> 23) & 0xFFu) == 0xFFu) x = 0.f;
    mf[i] = x;
}

// ---------------- edge pass B: exp + segment sum ----------------
__global__ void k_edge_exp(const float* __restrict__ el, const float* __restrict__ er,
                           const int* __restrict__ src, const int* __restrict__ dst,
                           const float* __restrict__ mf, float* __restrict__ s,
                           float* __restrict__ ee, int E, int H)
{
    int t = blockIdx.x * blockDim.x + threadIdx.x;
    if (t >= E * H) return;
    int e = t / H, h = t - e * H;
    float x = el[src[e] * H + h] + er[dst[e] * H + h];
    x = x > 0.f ? x : 0.2f * x;
    float v = __expf(x - mf[dst[e] * H + h]);
    ee[t] = v;
    atomicAdd(&s[dst[e] * H + h], v);
}

// ---------------- edge pass C: alpha-weighted scatter aggregate ----------------
// one wave per (edge, head); lane handles D/32 contiguous floats (float4 chunks)
__global__ __launch_bounds__(256) void k_edge_aggr(
    const float* __restrict__ Ht, const float* __restrict__ ee, const float* __restrict__ s,
    const int* __restrict__ src, const int* __restrict__ dst,
    float* __restrict__ out, int E, int H, int D)
{
    int gw   = blockIdx.x * (blockDim.x >> 5) + (threadIdx.x >> 5);
    int lane = threadIdx.x & 31;
    if (gw >= E * H) return;
    int e = gw / H, h = gw - e * H;
    int sN = src[e], dN = dst[e];
    float alpha = ee[gw] / (s[dN * H + h] + 1e-9f);
    int F = H * D;
    const float* rowS = Ht  + (size_t)sN * F + h * D;
    float*       rowD = out + (size_t)dN * F + h * D;
    int c = D >> 5;                 // 4 (D=128) or 8 (D=256) floats per lane
    int base = lane * c;
    for (int j = 0; j < c; j += 4) {
        float4 v = *(const float4*)(rowS + base + j);
        atomicAdd(rowD + base + j + 0, alpha * v.x);
        atomicAdd(rowD + base + j + 1, alpha * v.y);
        atomicAdd(rowD + base + j + 2, alpha * v.z);
        atomicAdd(rowD + base + j + 3, alpha * v.w);
    }
}

// ---------------- ELU + convert to f16 for layer-2 GEMM ----------------
__global__ void k_elu_cvt(const float* __restrict__ in, _Float16* __restrict__ out, int n) {
    int i = blockIdx.x * blockDim.x + threadIdx.x;
    if (i >= n) return;
    float x = in[i];
    float y = x > 0.f ? x : (__expf(x) - 1.0f);
    out[i] = (_Float16)y;
}

// ---------------------------------------------------------------------------
extern "C" void kernel_launch(void* const* d_in, const int* in_sizes, int n_in,
                              void* d_out, int out_size, void* d_ws, size_t ws_size,
                              hipStream_t stream)
{
    (void)in_sizes; (void)n_in; (void)out_size; (void)ws_size;
    const float* feat = (const float*)d_in[0];
    const int*   src1 = (const int*)  d_in[1];
    const int*   dst1 = (const int*)  d_in[2];
    const int*   src2 = (const int*)  d_in[3];
    const int*   dst2 = (const int*)  d_in[4];
    const float* W1   = (const float*)d_in[5];
    const float* al1  = (const float*)d_in[6];
    const float* ar1  = (const float*)d_in[7];
    const float* W2   = (const float*)d_in[8];
    const float* al2  = (const float*)d_in[9];
    const float* ar2  = (const float*)d_in[10];
    float* out = (float*)d_out;

    const int N = GAT_N, E = GAT_E;
    const int F1 = GAT_H1 * GAT_D1;     // 512
    const int F2 = GAT_H2 * GAT_D2;     // 256

    // workspace carve-out (256B aligned)
    char* ws = (char*)d_ws;
    size_t off = 0;
    auto carve = [&](size_t bytes) -> char* {
        char* p = ws + off;
        off += (bytes + 255) & ~(size_t)255;
        return p;
    };
    _Float16* Xh    = (_Float16*)carve((size_t)N * GAT_IND * 2);
    _Float16* W1h   = (_Float16*)carve((size_t)GAT_IND * F1 * 2);
    _Float16* W2h   = (_Float16*)carve((size_t)F1 * F2 * 2);
    float*    H1t   = (float*)   carve((size_t)N * F1 * 4);   // X @ W1
    float*    out1  = (float*)   carve((size_t)N * F1 * 4);   // layer1 aggregate
    _Float16* H1a   = (_Float16*)carve((size_t)N * F1 * 2);   // elu(out1) f16
    float*    H2t   = (float*)   carve((size_t)N * F2 * 4);   // h1 @ W2
    float*    el1   = (float*)   carve((size_t)N * GAT_H1 * 4);
    float*    er1   = (float*)   carve((size_t)N * GAT_H1 * 4);
    unsigned* m1u   = (unsigned*)carve((size_t)N * GAT_H1 * 4);
    float*    m1f   = (float*)   carve((size_t)N * GAT_H1 * 4);
    float*    s1    = (float*)   carve((size_t)N * GAT_H1 * 4);
    float*    ee1   = (float*)   carve((size_t)E * GAT_H1 * 4);
    float*    el2   = (float*)   carve((size_t)N * 4);
    float*    er2   = (float*)   carve((size_t)N * 4);
    unsigned* m2u   = (unsigned*)carve((size_t)N * 4);
    float*    m2f   = (float*)   carve((size_t)N * 4);
    float*    s2    = (float*)   carve((size_t)N * 4);
    float*    ee2   = (float*)   carve((size_t)E * 4);

    const int T = 256;
    auto blocks = [](int n, int t) { return (n + t - 1) / t; };

    // ---- precision conversion ----
    k_cvt_f16<<<blocks(N * GAT_IND, T), T, 0, stream>>>(feat, Xh, N * GAT_IND);
    k_cvt_f16<<<blocks(GAT_IND * F1, T), T, 0, stream>>>(W1, W1h, GAT_IND * F1);
    k_cvt_f16<<<blocks(F1 * F2, T), T, 0, stream>>>(W2, W2h, F1 * F2);

    // ================= layer 1 =================
    {
        dim3 g(blocks(N, BLK_M), F1 / BLK_N);
        k_gemm_f16_wmma<<<g, 256, 0, stream>>>(Xh, W1h, H1t, N, F1, GAT_IND);
    }
    k_node_attn<<<blocks(N * GAT_H1, T), T, 0, stream>>>(H1t, al1, ar1, el1, er1, N, GAT_H1, GAT_D1);

    k_fill_u32<<<blocks(N * GAT_H1, T), T, 0, stream>>>(m1u, 0u, N * GAT_H1);
    k_fill_u32<<<blocks(N * GAT_H1, T), T, 0, stream>>>((unsigned*)s1, 0u, N * GAT_H1);
    k_fill_u32<<<blocks(N * F1, T), T, 0, stream>>>((unsigned*)out1, 0u, N * F1);

    k_edge_max<<<blocks(E * GAT_H1, T), T, 0, stream>>>(el1, er1, src1, dst1, m1u, E, GAT_H1);
    k_m_fin<<<blocks(N * GAT_H1, T), T, 0, stream>>>(m1u, m1f, N * GAT_H1);
    k_edge_exp<<<blocks(E * GAT_H1, T), T, 0, stream>>>(el1, er1, src1, dst1, m1f, s1, ee1, E, GAT_H1);
    k_edge_aggr<<<blocks(E * GAT_H1 * 32, T), T, 0, stream>>>(H1t, ee1, s1, src1, dst1, out1, E, GAT_H1, GAT_D1);

    k_elu_cvt<<<blocks(N * F1, T), T, 0, stream>>>(out1, H1a, N * F1);

    // ================= layer 2 =================
    {
        dim3 g(blocks(N, BLK_M), F2 / BLK_N);
        k_gemm_f16_wmma<<<g, 256, 0, stream>>>(H1a, W2h, H2t, N, F2, F1);
    }
    k_node_attn<<<blocks(N * GAT_H2, T), T, 0, stream>>>(H2t, al2, ar2, el2, er2, N, GAT_H2, GAT_D2);

    k_fill_u32<<<blocks(N, T), T, 0, stream>>>(m2u, 0u, N);
    k_fill_u32<<<blocks(N, T), T, 0, stream>>>((unsigned*)s2, 0u, N);
    k_fill_u32<<<blocks(N * F2, T), T, 0, stream>>>((unsigned*)out, 0u, N * F2);

    k_edge_max<<<blocks(E, T), T, 0, stream>>>(el2, er2, src2, dst2, m2u, E, GAT_H2);
    k_m_fin<<<blocks(N, T), T, 0, stream>>>(m2u, m2f, N);
    k_edge_exp<<<blocks(E, T), T, 0, stream>>>(el2, er2, src2, dst2, m2f, s2, ee2, E, GAT_H2);
    k_edge_aggr<<<blocks(E * 32, T), T, 0, stream>>>(H2t, ee2, s2, src2, dst2, out, E, GAT_H2, GAT_D2);
}
